// RelMultiheadAttention_15470472200833
// MI455X (gfx1250) — matrix operand order repaired and checked by reading, not compiled, not run on hardware
//
#include <hip/hip_runtime.h>
#include <hip/hip_bf16.h>

// ---------- types ----------
typedef __attribute__((ext_vector_type(16))) __bf16    v16bf;
typedef __attribute__((ext_vector_type(8)))  float     v8f;
typedef __attribute__((ext_vector_type(8)))  unsigned  v8u;
typedef unsigned long long u64;

#define T_LEN 1024
#define S_LEN 1024
#define BATCH 8
#define NH 8
#define DH 32
#define DM 256
#define RELROWS 1023   // 2*MEM_LEN-1
#define RELMAX  1022   // 2*MEM_LEN-2
#define SCALE 0.17677669529663687f  // 1/sqrt(32)

// ---------- bf16 helpers ----------
__device__ __forceinline__ __bf16 f2bf(float f) { return (__bf16)f; }
__device__ __forceinline__ unsigned short bfbits(float f) {
  return __builtin_bit_cast(unsigned short, (__bf16)f);
}
__device__ __forceinline__ float bf2f(unsigned short h) {
  unsigned u = ((unsigned)h) << 16;
  return __builtin_bit_cast(float, u);
}
__device__ __forceinline__ unsigned pack2bf(float x, float y) {  // -> v_cvt_pk_bf16_f32
  return (unsigned)bfbits(x) | ((unsigned)bfbits(y) << 16);
}
// A-operand (16x32 bf16) K-pair base for VGPR v, lane-half hl (contiguous: k = 16*hl + 2*v)
__device__ __forceinline__ int a_kpair(int v, int hl) {
  return hl * 16 + v * 2;
}

// =====================================================================
// f32 -> bf16 packed convert-copy (n must be even)
// =====================================================================
__global__ void cvt_kernel(const float* __restrict__ src,
                           unsigned* __restrict__ dst, int n2) {
  int i = blockIdx.x * blockDim.x + threadIdx.x;   // over n/2 dwords
  if (i < n2) {
    float2 f = *(const float2*)(src + 2 * (size_t)i);
    dst[i] = pack2bf(f.x, f.y);
  }
}

// =====================================================================
// bf16 GEMM: out = A @ W^T (+ bias, x oscale), A [8192,256], W [256,256].
// MODE 0: bf16 out at [B,H,len,Dh]      (Q with oscale=1/sqrt(Dh), K)
// MODE 2: bf16 out at [B,H,Dh,len]      (V transposed for PV B-operand)
// MODE 3: f32 out at [row, col]         (final output projection)
// Block 256 thr = 8 waves (4 row-tiles x 2 col-tiles) -> 64x32 tile.
// =====================================================================
template <int MODE>
__global__ __launch_bounds__(256) void gemm_bf16_kernel(
    const unsigned short* __restrict__ Abf, const unsigned short* __restrict__ Wbf,
    const float* __restrict__ bias, void* __restrict__ outp, float oscale) {
  const int lane = threadIdx.x & 31;
  const int wave = threadIdx.x >> 5;
  const int hl = lane >> 4;
  const int ln = lane & 15;
  const int row0 = blockIdx.x * 64 + (wave & 3) * 16;
  const int col0 = blockIdx.y * 32 + (wave >> 2) * 16;

  v8f acc = {};
  for (int k0 = 0; k0 < DM; k0 += 32) {
    v8u au, bu;
    const unsigned short* arow = Abf + (size_t)(row0 + ln) * DM + k0;
    const unsigned short* brow = Wbf + (size_t)(col0 + ln) * DM + k0 + hl * 16;
#pragma unroll
    for (int v = 0; v < 8; ++v) {
      au[v] = *(const unsigned*)(arow + a_kpair(v, hl));
      bu[v] = *(const unsigned*)(brow + 2 * v);
    }
    acc = __builtin_amdgcn_wmma_f32_16x16x32_bf16(
        false, __builtin_bit_cast(v16bf, au), false, __builtin_bit_cast(v16bf, bu),
        (short)0, acc, false, false);
  }
  const int col = col0 + ln;
  const float bv = bias[col];
  const int h = col >> 5, dh = col & 31;
#pragma unroll
  for (int r = 0; r < 8; ++r) {
    int row = row0 + r + 8 * hl;          // row = t*B + b
    float val = (acc[r] + bv) * oscale;
    if (MODE == 3) {
      ((float*)outp)[(size_t)row * DM + col] = val;
    } else {
      int t = row >> 3, b = row & 7;
      size_t off;
      if (MODE == 2) off = ((size_t)((b * NH + h) * DH + dh)) * S_LEN + t;
      else           off = ((size_t)((b * NH + h) * T_LEN + t)) * DH + dh;
      ((unsigned short*)outp)[off] = bfbits(val);
    }
  }
}

// =====================================================================
// Fused TXL flash attention.
// Grid (T/128, B*H), block 256 = 8 waves; each wave owns 16 query rows.
// u/v bias rows are folded into two A-operands:
//   AC+UB = (sQ + su_h).K      AP+VB = (sQ + sv_h).rel[clip(t-s-M+511)]
// K/V chunk staged in LDS once per block. Per 32-wide S chunk:
//   2 WMMA (scores) + 3 WMMA (rel band) + 2 WMMA (P.V).
// =====================================================================
__global__ __launch_bounds__(256) void txl_attn_kernel(
    const unsigned short* __restrict__ Qb, const unsigned short* __restrict__ Kb,
    const unsigned short* __restrict__ Vt, const unsigned short* __restrict__ relbf,
    const float* __restrict__ uvec, const float* __restrict__ vvec,
    const int* __restrict__ mlen, unsigned short* __restrict__ attn_out) {
  __shared__ __align__(16) unsigned short kbuf[32 * DH];   // [s_local][dh]
  __shared__ __align__(16) unsigned short vbuf[DH * 32];   // [dh][s_local]
  __shared__ float qrel_s[8][16 * 49];                     // per-wave (AP+VB) band
  __shared__ __align__(4) unsigned short p_s[8][16 * 36];  // per-wave P (bf16)

  const int tid = threadIdx.x;
  const int lane = tid & 31;
  const int wave = tid >> 5;
  const int hl = lane >> 4;
  const int ln = lane & 15;
  const int bh = blockIdx.y;
  const int h = bh & (NH - 1);
  const int b = bh >> 3;
  const int t0 = blockIdx.x * 128 + wave * 16;
  const int M = mlen[0];

  float* qrel = qrel_s[wave];
  unsigned short* pbuf = p_s[wave];

  const unsigned short* Kbh = Kb + (size_t)bh * S_LEN * DH;
  const unsigned short* Vth = Vt + (size_t)bh * DH * S_LEN;

  // Q tile (pre-scaled by 1/sqrt(Dh)) + folded scaled u/v rows, A-layout.
  v16bf qau, qav;
  {
    const unsigned short* qrow = Qb + ((size_t)bh * T_LEN + (t0 + ln)) * DH;
    const float* urow = uvec + h * DH;
    const float* vrow = vvec + h * DH;
#pragma unroll
    for (int v = 0; v < 8; ++v) {
      int ka = a_kpair(v, hl);
      unsigned p = *(const unsigned*)(qrow + ka);
      float q0 = bf2f((unsigned short)(p & 0xffffu));
      float q1 = bf2f((unsigned short)(p >> 16));
      qau[2 * v]     = f2bf(q0 + SCALE * urow[ka]);
      qau[2 * v + 1] = f2bf(q1 + SCALE * urow[ka + 1]);
      qav[2 * v]     = f2bf(q0 + SCALE * vrow[ka]);
      qav[2 * v + 1] = f2bf(q1 + SCALE * vrow[ka + 1]);
    }
  }

  float mrow[8], lrow[8];
#pragma unroll
  for (int r = 0; r < 8; ++r) { mrow[r] = -1e30f; lrow[r] = 0.f; }
  v8f o0 = {}, o1 = {};

  for (int s0 = 0; s0 < S_LEN; s0 += 32) {
    // ---- cooperative staging of K/V chunk into LDS ----
    __syncthreads();
    ((u64*)kbuf)[tid] = ((const u64*)(Kbh + (size_t)s0 * DH))[tid];  // 2KB contiguous
    if (tid < 128) {
      int dh = tid >> 2, q = tid & 3;
      ((u64*)vbuf)[tid] = *((const u64*)(Vth + (size_t)dh * S_LEN + s0) + q);
    }
    // prefetch next chunk while this one computes (global_prefetch_b8)
    {
      int snext = s0 + 32 < S_LEN ? s0 + 32 : s0;
      __builtin_prefetch(Kbh + (size_t)snext * DH + tid * 8, 0, 1);
      if (tid < 32) __builtin_prefetch(Vth + (size_t)tid * S_LEN + snext, 0, 1);
    }
    __syncthreads();

    // ---- scores part 1: (sQ+su).K^T, two 16x16 tiles from LDS ----
    v8u k0u, k1u;
    {
      const unsigned short* k0p = kbuf + ln * DH + hl * 16;
      const unsigned short* k1p = kbuf + (16 + ln) * DH + hl * 16;
#pragma unroll
      for (int v = 0; v < 8; ++v) {
        k0u[v] = *(const unsigned*)(k0p + 2 * v);
        k1u[v] = *(const unsigned*)(k1p + 2 * v);
      }
    }
    v8f zc = {};
    v8f ac0 = __builtin_amdgcn_wmma_f32_16x16x32_bf16(
        false, qau, false, __builtin_bit_cast(v16bf, k0u), (short)0, zc, false, false);
    v8f ac1 = __builtin_amdgcn_wmma_f32_16x16x32_bf16(
        false, qau, false, __builtin_bit_cast(v16bf, k1u), (short)0, zc, false, false);

    // ---- scores part 2: (sQ+sv).rel band, 3 WMMAs -> LDS ----
    const int jbase = t0 - s0 - M + 480;   // (t-s-M+511) - 31
#pragma unroll
    for (int tile = 0; tile < 3; ++tile) {
      v8u ru;
      int jr = jbase + tile * 16 + ln;
      jr = jr < 0 ? 0 : (jr > RELMAX ? RELMAX : jr);
      const unsigned short* rrow = relbf + (size_t)jr * DH + hl * 16;
#pragma unroll
      for (int v = 0; v < 8; ++v) ru[v] = *(const unsigned*)(rrow + 2 * v);
      v8f qr = __builtin_amdgcn_wmma_f32_16x16x32_bf16(
          false, qav, false, __builtin_bit_cast(v16bf, ru), (short)0, zc, false, false);
#pragma unroll
      for (int r = 0; r < 8; ++r)
        qrel[(r + 8 * hl) * 49 + tile * 16 + ln] = qr[r];
    }
    asm volatile("s_wait_dscnt 0x0" ::: "memory");

    // ---- assemble scores: AC' + diagonal gather of (AP+VB) band ----
    float sc[2][8];
#pragma unroll
    for (int tile = 0; tile < 2; ++tile) {
      const int n = tile * 16 + ln;
      const v8f& acv = tile ? ac1 : ac0;
#pragma unroll
      for (int r = 0; r < 8; ++r) {
        int m = r + 8 * hl;
        sc[tile][r] = acv[r] + qrel[m * 49 + (m - n + 31)];
      }
    }

    // ---- online softmax (rows live on 16-lane half-groups) ----
#pragma unroll
    for (int r = 0; r < 8; ++r) {
      float vmax = fmaxf(sc[0][r], sc[1][r]);
#pragma unroll
      for (int mask = 1; mask <= 8; mask <<= 1)
        vmax = fmaxf(vmax, __shfl_xor(vmax, mask, 32));
      float mnew = fmaxf(mrow[r], vmax);
      float alpha = __expf(mrow[r] - mnew);
      float p0 = __expf(sc[0][r] - mnew);
      float p1 = __expf(sc[1][r] - mnew);
      float psum = p0 + p1;
#pragma unroll
      for (int mask = 1; mask <= 8; mask <<= 1)
        psum += __shfl_xor(psum, mask, 32);
      lrow[r] = lrow[r] * alpha + psum;
      mrow[r] = mnew;
      o0[r] *= alpha; o1[r] *= alpha;
      int m = r + 8 * hl;
      pbuf[m * 36 + ln] = bfbits(p0);        // P parked directly as bf16
      pbuf[m * 36 + 16 + ln] = bfbits(p1);
    }
    asm volatile("s_wait_dscnt 0x0" ::: "memory");

    // ---- P -> A-operand layout (dword LDS reads), V tiles from LDS ----
    v8u pu, v0u, v1u;
    {
      const unsigned short* prow = pbuf + ln * 36;
      const unsigned short* v0p = vbuf + ln * 32 + hl * 16;
      const unsigned short* v1p = vbuf + (16 + ln) * 32 + hl * 16;
#pragma unroll
      for (int v = 0; v < 8; ++v) {
        pu[v] = *(const unsigned*)(prow + a_kpair(v, hl));
        v0u[v] = *(const unsigned*)(v0p + 2 * v);
        v1u[v] = *(const unsigned*)(v1p + 2 * v);
      }
    }
    o0 = __builtin_amdgcn_wmma_f32_16x16x32_bf16(
        false, __builtin_bit_cast(v16bf, pu), false, __builtin_bit_cast(v16bf, v0u),
        (short)0, o0, false, false);
    o1 = __builtin_amdgcn_wmma_f32_16x16x32_bf16(
        false, __builtin_bit_cast(v16bf, pu), false, __builtin_bit_cast(v16bf, v1u),
        (short)0, o1, false, false);
  }

  // ---- finalize: O / l, store bf16 at [t*B+b, h*32+dh] for output GEMM ----
#pragma unroll
  for (int r = 0; r < 8; ++r) {
    int m = r + 8 * hl;
    int t = t0 + m;
    float inv = 1.0f / lrow[r];
    unsigned short* orow = attn_out + ((size_t)t * BATCH + b) * DM + h * DH;
    orow[ln] = bfbits(o0[r] * inv);
    orow[16 + ln] = bfbits(o1[r] * inv);
  }
}

// =====================================================================
// launcher
// =====================================================================
extern "C" void kernel_launch(void* const* d_in, const int* in_sizes, int n_in,
                              void* d_out, int out_size, void* d_ws, size_t ws_size,
                              hipStream_t stream) {
  const float* query = (const float*)d_in[0];
  const float* key   = (const float*)d_in[1];
  const float* value = (const float*)d_in[2];
  const float* Wq = (const float*)d_in[3];
  const float* bq = (const float*)d_in[4];
  const float* Wk = (const float*)d_in[5];
  const float* bk = (const float*)d_in[6];
  const float* Wv = (const float*)d_in[7];
  const float* bv = (const float*)d_in[8];
  const float* Wo = (const float*)d_in[9];
  const float* bo = (const float*)d_in[10];
  const float* rel = (const float*)d_in[11];
  const float* u   = (const float*)d_in[12];
  const float* vv  = (const float*)d_in[13];
  const int*   mlen = (const int*)d_in[14];
  float* out = (float*)d_out;

  // workspace carve-up (all bf16/ushort, ~29 MB)
  const size_t NE = (size_t)BATCH * NH * T_LEN * DH;  // 2,097,152 (= 8192*256)
  const size_t NW = (size_t)DM * DM;                  // 65,536
  unsigned short* Qb = (unsigned short*)d_ws;
  unsigned short* Kb  = Qb + NE;
  unsigned short* Vt  = Kb + NE;
  unsigned short* Ao  = Vt + NE;
  unsigned short* Xq  = Ao + NE;
  unsigned short* Xk  = Xq + NE;
  unsigned short* Xv  = Xk + NE;
  unsigned short* relbf = Xv + NE;
  unsigned short* Wqb = relbf + (size_t)RELROWS * DH;
  unsigned short* Wkb = Wqb + NW;
  unsigned short* Wvb = Wkb + NW;
  unsigned short* Wob = Wvb + NW;

  // --- pre-convert all GEMM operands to bf16 (packed cvt) ---
  const int half_NE = (int)(NE / 2), half_NW = (int)(NW / 2);
  cvt_kernel<<<(half_NE + 255) / 256, 256, 0, stream>>>(query, (unsigned*)Xq, half_NE);
  cvt_kernel<<<(half_NE + 255) / 256, 256, 0, stream>>>(key,   (unsigned*)Xk, half_NE);
  cvt_kernel<<<(half_NE + 255) / 256, 256, 0, stream>>>(value, (unsigned*)Xv, half_NE);
  cvt_kernel<<<(half_NW + 255) / 256, 256, 0, stream>>>(Wq, (unsigned*)Wqb, half_NW);
  cvt_kernel<<<(half_NW + 255) / 256, 256, 0, stream>>>(Wk, (unsigned*)Wkb, half_NW);
  cvt_kernel<<<(half_NW + 255) / 256, 256, 0, stream>>>(Wv, (unsigned*)Wvb, half_NW);
  cvt_kernel<<<(half_NW + 255) / 256, 256, 0, stream>>>(Wo, (unsigned*)Wob, half_NW);
  const int half_rel = RELROWS * DH / 2;
  cvt_kernel<<<(half_rel + 255) / 256, 256, 0, stream>>>(rel, (unsigned*)relbf, half_rel);

  dim3 gblk(256);
  dim3 ggrid(T_LEN * BATCH / 64, DM / 32);   // (128, 8)

  gemm_bf16_kernel<0><<<ggrid, gblk, 0, stream>>>(Xq, Wqb, bq, Qb, SCALE);
  gemm_bf16_kernel<0><<<ggrid, gblk, 0, stream>>>(Xk, Wkb, bk, Kb, 1.0f);
  gemm_bf16_kernel<2><<<ggrid, gblk, 0, stream>>>(Xv, Wvb, bv, Vt, 1.0f);

  dim3 agrid(T_LEN / 128, BATCH * NH);       // (8, 64)
  txl_attn_kernel<<<agrid, gblk, 0, stream>>>(Qb, Kb, Vt, relbf, u, vv, mlen, Ao);

  gemm_bf16_kernel<3><<<ggrid, gblk, 0, stream>>>(Ao, Wob, bo, out, 1.0f);
}
